// ReconstructionNet_53704271069765
// MI455X (gfx1250) — matrix operand order repaired and checked by reading, not compiled, not run on hardware
//
#include <hip/hip_runtime.h>

// ---------------------------------------------------------------------------
// ReconstructionNet (DGCNN encoder + FoldNet decoder) for MI455X / gfx1250.
// B=8, N=2048, K=20, M=2025, FD=512.
//  - All GEMMs on v_wmma_f32_16x16x32_f16 (f16 in, fp32 accumulate),
//    2x4 register blocking (8 WMMAs per K-step, 0.75 KB LDS per WMMA).
//  - Weight tiles staged LDS-side with GLOBAL_LOAD_ASYNC_TO_LDS_B128
//    (ASYNCcnt + s_wait_asynccnt), no VGPR round-trip.
//  - KNN kept in fp32 VALU (register-resident distances, 20 argmax passes).
// Assumes ws_size >= ~110 MB.
// ---------------------------------------------------------------------------

#define DEVFN __device__ __forceinline__

constexpr int cB = 8, cN = 2048, cK = 20, cM = 2025, cFD = 512;

typedef __attribute__((ext_vector_type(16))) _Float16 v16h;
typedef __attribute__((ext_vector_type(8)))  float    v8f;

union FragU { unsigned int u[8]; v16h h; };

// A fragment: LDS tile is 16 rows x K f16, row-major, strideH halves.
// lane = hi*16 + r ; VGPR v holds K-pair p = (v>=4)*8 + hi*4 + (v&3).
DEVFN v16h frag_a(const _Float16* tile, int strideH, int kt, int lane) {
  const unsigned int* row =
      (const unsigned int*)(tile + (lane & 15) * strideH + kt * 32);
  int hi = (lane >> 4) & 1;
  FragU f;
#pragma unroll
  for (int v = 0; v < 8; ++v) {
    int p = ((v & 4) << 1) + hi * 4 + (v & 3);
    f.u[v] = row[p];
  }
  return f.h;
}

// B fragment from B^T tile (N rows x K f16, row-major): lane n = lane&15,
// VGPR v holds K-pair p = (v>=4)*8 + hi*4 + (v&3).
DEVFN v16h frag_b(const _Float16* tile, int strideH, int kt, int lane) {
  const unsigned int* row =
      (const unsigned int*)(tile + (lane & 15) * strideH + kt * 32);
  int hi = (lane >> 4) & 1;
  FragU f;
#pragma unroll
  for (int v = 0; v < 8; ++v) f.u[v] = row[((v & 4) << 1) + hi * 4 + (v & 3)];
  return f.h;
}

DEVFN v8f wmma16(v16h a, v16h b, v8f c) {
  return __builtin_amdgcn_wmma_f32_16x16x32_f16(false, a, false, b, (short)0,
                                                c, false, false);
}

// Async global -> LDS copy of one 16B chunk per active lane (ASYNCcnt).
DEVFN void async_b128(unsigned long long gbase, unsigned gByteOff, void* ldsPtr) {
  unsigned ldsOff = (unsigned)(unsigned long long)ldsPtr;  // addr[31:0] = LDS offset
  asm volatile("global_load_async_to_lds_b128 %0, %1, %2"
               :
               : "v"(ldsOff), "v"(gByteOff), "s"(gbase)
               : "memory");
}
DEVFN void wait_async0() { asm volatile("s_wait_asynccnt 0" ::: "memory"); }

DEVFN void atomicMaxF(float* addr, float val) {
  unsigned int* ua = (unsigned int*)addr;
  unsigned int old = *ua;
  while (__uint_as_float(old) < val) {
    unsigned int assumed = old;
    old = atomicCAS(ua, assumed, __float_as_uint(val));
    if (old == assumed) break;
  }
}

// ---------------------------------------------------------------------------
// Utility kernels
// ---------------------------------------------------------------------------
__global__ void init_kernel(float* p, int n, float v) {
  int i = blockIdx.x * blockDim.x + threadIdx.x;
  if (i < n) p[i] = v;
}

// fp32 -> f16 with row/col zero padding. out is (outRows x kpad).
__global__ void pad_convert(const float* __restrict__ in, _Float16* __restrict__ out,
                            int outRows, int inRows, int kin, int kpad) {
  int i = blockIdx.x * blockDim.x + threadIdx.x;
  int total = outRows * kpad;
  if (i >= total) return;
  int r = i / kpad, c = i % kpad;
  float v = (r < inRows && c < kin) ? in[r * kin + c] : 0.f;
  out[i] = (_Float16)v;
}

// xx[b,n] = sum_c x[b,n,c]^2   (x has row stride xs)
__global__ void xx_kernel(const float* __restrict__ x, int xs, int C,
                          float* __restrict__ xx, int total) {
  int i = blockIdx.x * blockDim.x + threadIdx.x;
  if (i >= total) return;
  const float* p = x + (long)i * xs;
  float s = 0.f;
  for (int c = 0; c < C; ++c) s += p[c] * p[c];
  xx[i] = s;
}

// ---------------------------------------------------------------------------
// KNN: per wave one row.  Distances (2048) in 64 VGPRs/lane, 20 iterative
// wave-wide argmax passes.  Column features tiled through LDS (shared by the
// 8 waves of the block; all rows of a block belong to one batch).
// ---------------------------------------------------------------------------
template <int C>
__launch_bounds__(256)
__global__ void knn_kernel(const float* __restrict__ x, int xs,
                           const float* __restrict__ xx, int* __restrict__ idxOut) {
  extern __shared__ float smf[];
  float* colfT = smf;                 // [C][64] transposed: bank-conflict-free
  float* colxx = colfT + C * 64;      // [64]
  float* rowf  = colxx + 64;          // [8][C]
  const int tid = threadIdx.x, lane = tid & 31, wave = tid >> 5;
  const int b   = blockIdx.x / (cN / 8);
  const int row = (blockIdx.x % (cN / 8)) * 8 + wave;

  for (int c = lane; c < C; c += 32)
    rowf[wave * C + c] = x[(long)(b * cN + row) * xs + c];
  const float rxx = xx[b * cN + row];

  float d[cN / 32];
  __syncthreads();
  for (int t = 0; t < cN / 64; ++t) {
    for (int e = tid; e < C * 64; e += 256) {
      int c = e >> 6, jj = e & 63;
      colfT[e] = x[(long)(b * cN + t * 64 + jj) * xs + c];
    }
    if (tid < 64) colxx[tid] = xx[b * cN + t * 64 + tid];
    __syncthreads();
    float dot0 = 0.f, dot1 = 0.f;
    for (int c = 0; c < C; ++c) {
      float rf = rowf[wave * C + c];
      dot0 += rf * colfT[c * 64 + lane];
      dot1 += rf * colfT[c * 64 + 32 + lane];
    }
    d[2 * t]     = 2.f * dot0 - rxx - colxx[lane];
    d[2 * t + 1] = 2.f * dot1 - rxx - colxx[32 + lane];
    __syncthreads();
  }
  for (int s = 0; s < cK; ++s) {
    float bv = -3.4e38f;
    int bj = 0;
#pragma unroll
    for (int j = 0; j < cN / 32; ++j)
      if (d[j] > bv) { bv = d[j]; bj = j; }
    int bc = bj * 32 + lane;
    for (int off = 16; off; off >>= 1) {
      float ov = __shfl_xor(bv, off, 32);
      int   oc = __shfl_xor(bc, off, 32);
      if (ov > bv || (ov == bv && oc < bc)) { bv = ov; bc = oc; }
    }
    if ((bc & 31) == lane) d[bc >> 5] = -3.4e38f;
    if (lane == 0) idxOut[((long)(b * cN) + row) * cK + s] = bc;
  }
}

// ---------------------------------------------------------------------------
// Fused EdgeConv: per wave one point.  A = [nb-ctr, ctr] (32 x KP f16, rows
// 20..31 zero) in LDS.  2 M-tiles x 4 N-tiles register blocking: per K-step
// 2 A-frags + 4 B-frags -> 8 WMMAs.  Weight panel (64 x KP) staged with
// async global->LDS b128.  Epilogue: scale+bias+LeakyReLU, max over k.
// ---------------------------------------------------------------------------
template <int CIN, int COUT>
__launch_bounds__(256)
__global__ void edgeconv_kernel(const float* __restrict__ x, int xs,
                                const int* __restrict__ idx,
                                const _Float16* __restrict__ w,   // (COUT, KP)
                                const float* __restrict__ scl,
                                const float* __restrict__ bias,
                                float* __restrict__ out, int outW, int outOff) {
  constexpr int KP = (2 * CIN + 31) & ~31;
  constexpr int NB = 4;                       // N-tiles per panel (64 outputs)
  extern __shared__ _Float16 smh_e[];
  const int NW = 8;
  _Float16* ldsA = smh_e;                     // NW * 32 * KP
  _Float16* ldsB = smh_e + NW * 32 * KP;      // 64 * KP
  const int tid = threadIdx.x, lane = tid & 31, wave = tid >> 5;
  const int p = blockIdx.x * NW + wave;
  const int b = p / cN, n = p % cN;
  _Float16* myA = ldsA + wave * 32 * KP;

  {  // build A: one lane per row k
    int k = lane;
    _Float16* row = myA + k * KP;
    if (k < cK) {
      int nbr = idx[(long)p * cK + k];
      const float* xn = x + (long)(b * cN + n) * xs;
      const float* xb = x + (long)(b * cN + nbr) * xs;
      for (int c = 0; c < CIN; ++c) {
        float cv = xn[c];
        row[c]       = (_Float16)(xb[c] - cv);
        row[CIN + c] = (_Float16)cv;
      }
      for (int c = 2 * CIN; c < KP; ++c) row[c] = (_Float16)0.f;
    } else {
      for (int c = 0; c < KP; ++c) row[c] = (_Float16)0.f;
    }
  }
  __syncthreads();

  for (int ng = 0; ng < COUT / (16 * NB); ++ng) {
    // async-stage 64 x KP weight panel (chunks = 2*NB*KP, exact multiple of 256)
    {
      unsigned long long gb = (unsigned long long)(w + (long)ng * (16 * NB) * KP);
      char* dst = (char*)ldsB;
      for (int e = tid; e < 2 * NB * KP; e += 256)
        async_b128(gb, (unsigned)(e * 16), dst + e * 16);
      wait_async0();
    }
    __syncthreads();

    v8f acc0[NB], acc1[NB];
#pragma unroll
    for (int j = 0; j < NB; ++j) {
      acc0[j] = v8f{0.f,0.f,0.f,0.f,0.f,0.f,0.f,0.f};
      acc1[j] = v8f{0.f,0.f,0.f,0.f,0.f,0.f,0.f,0.f};
    }
#pragma unroll
    for (int kt = 0; kt < KP / 32; ++kt) {
      v16h a0 = frag_a(myA, KP, kt, lane);
      v16h a1 = frag_a(myA + 16 * KP, KP, kt, lane);
#pragma unroll
      for (int j = 0; j < NB; ++j) {
        v16h bf = frag_b(ldsB + j * 16 * KP, KP, kt, lane);
        acc0[j] = wmma16(a0, bf, acc0[j]);
        acc1[j] = wmma16(a1, bf, acc1[j]);
      }
    }
    const int nl = lane & 15, hi = lane >> 4;
#pragma unroll
    for (int j = 0; j < NB; ++j) {
      const int o = (ng * NB + j) * 16 + nl;
      const float sv = scl[o], bv = bias[o];
      float mx = -3.4e38f;
#pragma unroll
      for (int v = 0; v < 8; ++v) {
        float h0 = acc0[j][v] * sv + bv;  h0 = h0 > 0.f ? h0 : 0.2f * h0;
        mx = fmaxf(mx, h0);                     // m = v + 8*hi (< 16, valid)
        int m1 = 16 + v + 8 * hi;
        if (m1 < cK) {
          float h1 = acc1[j][v] * sv + bv;  h1 = h1 > 0.f ? h1 : 0.2f * h1;
          mx = fmaxf(mx, h1);
        }
      }
      mx = fmaxf(mx, __shfl_xor(mx, 16, 32));
      if (lane < 16) out[(long)p * outW + outOff + o] = mx;
    }
    __syncthreads();
  }
}

// ---------------------------------------------------------------------------
// Generic WMMA GEMM: out = epilogue(A(f32, Arows x KA) * Bw(f16, O x KA)^T).
// Block: 4 waves x 32 rows = 128 rows; gridDim.y selects the 16*NB-wide
// output panel.  Per K-step: 2 A-frags + NB B-frags -> 2*NB WMMAs.
// EPI 0: leaky(acc*s[o]+b[o], 0.2) atomically max-reduced into out[(row/rpb)*O+o]
// EPI 1: out[row*outW+o] = relu(acc + bias[o])
// EPI 2: out[row*outW+o] = acc + bias[o]   (only o < outW)
// ---------------------------------------------------------------------------
template <int KA, int O, int NB, int EPI>
__launch_bounds__(128)
__global__ void gemm_kernel(const float* __restrict__ A, int Arows, int rpb,
                            const _Float16* __restrict__ Bw,
                            const float* __restrict__ scl,
                            const float* __restrict__ bias,
                            float* __restrict__ out, int outW) {
  extern __shared__ _Float16 smh_g[];
  const int NW = 4;
  _Float16* ldsA = smh_g;                     // NW * 32 * KA
  _Float16* ldsB = smh_g + NW * 32 * KA;      // 16*NB * KA
  const int tid = threadIdx.x, lane = tid & 31, wave = tid >> 5;
  const int rowBase = blockIdx.x * (NW * 32) + wave * 32;
  const int ng = blockIdx.y;
  _Float16* myA = ldsA + wave * 32 * KA;

  for (int r = 0; r < 32; ++r) {
    int row = rowBase + r;
    _Float16* dst = myA + r * KA;
    if (row < Arows) {
      const float* src = A + (long)row * KA;
      for (int c = lane; c < KA; c += 32) dst[c] = (_Float16)src[c];
    } else {
      for (int c = lane; c < KA; c += 32) dst[c] = (_Float16)0.f;
    }
  }
  // async-stage 16*NB x KA weight panel (chunks = 2*NB*KA, multiple of 128)
  {
    unsigned long long gb = (unsigned long long)(Bw + (long)ng * (16 * NB) * KA);
    char* dst = (char*)ldsB;
    for (int e = tid; e < 2 * NB * KA; e += 128)
      async_b128(gb, (unsigned)(e * 16), dst + e * 16);
    wait_async0();
  }
  __syncthreads();

  v8f acc0[NB], acc1[NB];
#pragma unroll
  for (int j = 0; j < NB; ++j) {
    acc0[j] = v8f{0.f,0.f,0.f,0.f,0.f,0.f,0.f,0.f};
    acc1[j] = v8f{0.f,0.f,0.f,0.f,0.f,0.f,0.f,0.f};
  }
#pragma unroll
  for (int kt = 0; kt < KA / 32; ++kt) {
    v16h a0 = frag_a(myA, KA, kt, lane);
    v16h a1 = frag_a(myA + 16 * KA, KA, kt, lane);
#pragma unroll
    for (int j = 0; j < NB; ++j) {
      v16h bf = frag_b(ldsB + j * 16 * KA, KA, kt, lane);
      acc0[j] = wmma16(a0, bf, acc0[j]);
      acc1[j] = wmma16(a1, bf, acc1[j]);
    }
  }
  const int nl = lane & 15, hi = lane >> 4;
#pragma unroll
  for (int mt = 0; mt < 2; ++mt) {
#pragma unroll
    for (int j = 0; j < NB; ++j) {
      const int o = (ng * NB + j) * 16 + nl;
#pragma unroll
      for (int v = 0; v < 8; ++v) {
        int row = rowBase + mt * 16 + v + 8 * hi;
        if (row < Arows) {
          float val = (mt == 0) ? acc0[j][v] : acc1[j][v];
          if constexpr (EPI == 0) {
            val = val * scl[o] + bias[o];
            val = val > 0.f ? val : 0.2f * val;
            atomicMaxF(&out[(row / rpb) * O + o], val);
          } else if constexpr (EPI == 1) {
            val += bias[o];
            out[(long)row * outW + o] = val > 0.f ? val : 0.f;
          } else {
            if (o < outW) out[(long)row * outW + o] = val + bias[o];
          }
        }
      }
    }
  }
}

// ---------------------------------------------------------------------------
// Decoder A-row builder: A[row] = [feat[b] (512) | extra (2 or 3) | zeros]
// extra == nullptr -> plane grid (meshgrid ij of linspace(-0.3, 0.3, 45)).
// ---------------------------------------------------------------------------
__global__ void build_fold_A(const float* __restrict__ feat,
                             const float* __restrict__ extra, int extraDim,
                             float* __restrict__ A, int rowsTotal) {
  const int KA = 544;
  long i = (long)blockIdx.x * blockDim.x + threadIdx.x;
  long total = (long)rowsTotal * KA;
  if (i >= total) return;
  int c = (int)(i % KA);
  long row = i / KA;
  int b = (int)(row / cM), m = (int)(row % cM);
  float v;
  if (c < cFD) {
    v = feat[b * cFD + c];
  } else if (c < cFD + extraDim) {
    if (extra) {
      v = extra[row * extraDim + (c - cFD)];
    } else {
      int gi = m / 45, gj = m % 45;
      int g = (c - cFD) == 0 ? gi : gj;
      v = -0.3f + 0.6f * (float)g / 44.f;
    }
  } else {
    v = 0.f;
  }
  A[i] = v;
}

// ---------------------------------------------------------------------------
// Host launch
// ---------------------------------------------------------------------------
static inline int cdiv(long a, long b) { return (int)((a + b - 1) / b); }

extern "C" void kernel_launch(void* const* d_in, const int* in_sizes, int n_in,
                              void* d_out, int out_size, void* d_ws, size_t ws_size,
                              hipStream_t stream) {
  (void)in_sizes; (void)n_in; (void)out_size; (void)ws_size;
  const float* points = (const float*)d_in[0];
  const float* w1 = (const float*)d_in[1];  const float* s1 = (const float*)d_in[2];  const float* b1 = (const float*)d_in[3];
  const float* w2 = (const float*)d_in[4];  const float* s2 = (const float*)d_in[5];  const float* b2 = (const float*)d_in[6];
  const float* w3 = (const float*)d_in[7];  const float* s3 = (const float*)d_in[8];  const float* b3 = (const float*)d_in[9];
  const float* w4 = (const float*)d_in[10]; const float* s4 = (const float*)d_in[11]; const float* b4 = (const float*)d_in[12];
  const float* w5 = (const float*)d_in[13]; const float* s5 = (const float*)d_in[14]; const float* b5 = (const float*)d_in[15];
  const float* f1w1 = (const float*)d_in[16]; const float* f1b1 = (const float*)d_in[17];
  const float* f1w2 = (const float*)d_in[18]; const float* f1b2 = (const float*)d_in[19];
  const float* f1w3 = (const float*)d_in[20]; const float* f1b3 = (const float*)d_in[21];
  const float* f2w1 = (const float*)d_in[22]; const float* f2b1 = (const float*)d_in[23];
  const float* f2w2 = (const float*)d_in[24]; const float* f2b2 = (const float*)d_in[25];
  const float* f2w3 = (const float*)d_in[26]; const float* f2b3 = (const float*)d_in[27];
  float* outp = (float*)d_out;

  // ---- workspace carve ----
  char* wp = (char*)d_ws;
  auto carve = [&](size_t bytes) -> void* {
    void* r = (void*)wp;
    wp += (bytes + 255) & ~(size_t)255;
    return r;
  };
  float* xx   = (float*)carve((size_t)cB * cN * 4);
  int*   idx  = (int*)carve((size_t)cB * cN * cK * 4);
  float* xc   = (float*)carve((size_t)cB * cN * 512 * 4);
  float* feat = (float*)carve((size_t)cB * 512 * 4);
  float* bufA = (float*)carve((size_t)cB * cM * 544 * 4);
  float* bufB = (float*)carve((size_t)cB * cM * 544 * 4);
  float* fr1  = (float*)carve((size_t)cB * cM * 3 * 4);
  _Float16* w1p   = (_Float16*)carve((size_t)64 * 32 * 2);
  _Float16* w2p   = (_Float16*)carve((size_t)64 * 128 * 2);
  _Float16* w3p   = (_Float16*)carve((size_t)128 * 128 * 2);
  _Float16* w4p   = (_Float16*)carve((size_t)256 * 256 * 2);
  _Float16* w5p   = (_Float16*)carve((size_t)512 * 512 * 2);
  _Float16* f1w1p = (_Float16*)carve((size_t)512 * 544 * 2);
  _Float16* f1w2p = (_Float16*)carve((size_t)512 * 512 * 2);
  _Float16* f1w3p = (_Float16*)carve((size_t)16 * 512 * 2);
  _Float16* f2w1p = (_Float16*)carve((size_t)512 * 544 * 2);
  _Float16* f2w2p = (_Float16*)carve((size_t)512 * 512 * 2);
  _Float16* f2w3p = (_Float16*)carve((size_t)16 * 512 * 2);

  // ---- raise dynamic LDS limits for big-LDS kernels ----
  auto setlds = [](const void* f, int sz) {
    hipFuncSetAttribute(f, hipFuncAttributeMaxDynamicSharedMemorySize, sz);
  };
  setlds(reinterpret_cast<const void*>(&edgeconv_kernel<64, 64>),    81920);
  setlds(reinterpret_cast<const void*>(&edgeconv_kernel<64, 128>),   81920);
  setlds(reinterpret_cast<const void*>(&edgeconv_kernel<128, 256>), 163840);
  setlds(reinterpret_cast<const void*>(&gemm_kernel<512, 512, 4, 0>), 196608);
  setlds(reinterpret_cast<const void*>(&gemm_kernel<512, 512, 4, 1>), 196608);
  setlds(reinterpret_cast<const void*>(&gemm_kernel<544, 512, 4, 1>), 208896);
  setlds(reinterpret_cast<const void*>(&gemm_kernel<512, 16, 1, 2>),  147456);

  // ---- weight conversion (f32 -> padded f16) ----
  auto cvt = [&](const float* in, _Float16* out, int outRows, int inRows, int kin, int kpad) {
    int total = outRows * kpad;
    pad_convert<<<cdiv(total, 256), 256, 0, stream>>>(in, out, outRows, inRows, kin, kpad);
  };
  cvt(w1, w1p, 64, 64, 6, 32);
  cvt(w2, w2p, 64, 64, 128, 128);
  cvt(w3, w3p, 128, 128, 128, 128);
  cvt(w4, w4p, 256, 256, 256, 256);
  cvt(w5, w5p, 512, 512, 512, 512);
  cvt(f1w1, f1w1p, 512, 512, 514, 544);
  cvt(f1w2, f1w2p, 512, 512, 512, 512);
  cvt(f1w3, f1w3p, 16, 3, 512, 512);
  cvt(f2w1, f2w1p, 512, 512, 515, 544);
  cvt(f2w2, f2w2p, 512, 512, 512, 512);
  cvt(f2w3, f2w3p, 16, 3, 512, 512);

  const int totPts = cB * cN;
  auto knnLds = [](int C) { return (size_t)(C * 64 + 64 + 8 * C) * 4; };
  auto ecLds  = [](int KP) { return (size_t)(8 * 32 * KP + 64 * KP) * 2; };
  auto gmLds  = [](int KA, int NB) { return (size_t)(4 * 32 * KA + 16 * NB * KA) * 2; };

  // ---- EdgeConv layer 1 (3 -> 64) ----
  xx_kernel<<<cdiv(totPts, 256), 256, 0, stream>>>(points, 3, 3, xx, totPts);
  knn_kernel<3><<<totPts / 8, 256, knnLds(3), stream>>>(points, 3, xx, idx);
  edgeconv_kernel<3, 64><<<totPts / 8, 256, ecLds(32), stream>>>(
      points, 3, idx, w1p, s1, b1, xc, 512, 0);

  // ---- EdgeConv layer 2 (64 -> 64) ----
  xx_kernel<<<cdiv(totPts, 256), 256, 0, stream>>>(xc, 512, 64, xx, totPts);
  knn_kernel<64><<<totPts / 8, 256, knnLds(64), stream>>>(xc, 512, xx, idx);
  edgeconv_kernel<64, 64><<<totPts / 8, 256, ecLds(128), stream>>>(
      xc, 512, idx, w2p, s2, b2, xc, 512, 64);

  // ---- EdgeConv layer 3 (64 -> 128) ----
  xx_kernel<<<cdiv(totPts, 256), 256, 0, stream>>>(xc + 64, 512, 64, xx, totPts);
  knn_kernel<64><<<totPts / 8, 256, knnLds(64), stream>>>(xc + 64, 512, xx, idx);
  edgeconv_kernel<64, 128><<<totPts / 8, 256, ecLds(128), stream>>>(
      xc + 64, 512, idx, w3p, s3, b3, xc, 512, 128);

  // ---- EdgeConv layer 4 (128 -> 256) ----
  xx_kernel<<<cdiv(totPts, 256), 256, 0, stream>>>(xc + 128, 512, 128, xx, totPts);
  knn_kernel<128><<<totPts / 8, 256, knnLds(128), stream>>>(xc + 128, 512, xx, idx);
  edgeconv_kernel<128, 256><<<totPts / 8, 256, ecLds(256), stream>>>(
      xc + 128, 512, idx, w4p, s4, b4, xc, 512, 256);

  // ---- conv5 (512->512) + global max-pool -> feat (B,512) ----
  init_kernel<<<cdiv(cB * 512, 256), 256, 0, stream>>>(feat, cB * 512, -3.4e38f);
  gemm_kernel<512, 512, 4, 0>
      <<<dim3(cdiv(totPts, 128), 8), 128, gmLds(512, 4), stream>>>(
      xc, totPts, cN, w5p, s5, b5, feat, 512);

  // ---- FoldNet decoder ----
  const int rowsD = cB * cM;  // 16200
  const int gD = cdiv(rowsD, 128);
  // fold 1: cat(feat, grid) -> 512 -> 512 -> 3
  build_fold_A<<<cdiv((long)rowsD * 544, 256), 256, 0, stream>>>(feat, nullptr, 2, bufA, rowsD);
  gemm_kernel<544, 512, 4, 1><<<dim3(gD, 8), 128, gmLds(544, 4), stream>>>(
      bufA, rowsD, cM, f1w1p, nullptr, f1b1, bufB, 512);
  gemm_kernel<512, 512, 4, 1><<<dim3(gD, 8), 128, gmLds(512, 4), stream>>>(
      bufB, rowsD, cM, f1w2p, nullptr, f1b2, bufA, 512);
  gemm_kernel<512, 16, 1, 2><<<dim3(gD, 1), 128, gmLds(512, 1), stream>>>(
      bufA, rowsD, cM, f1w3p, nullptr, f1b3, fr1, 3);
  // fold 2: cat(feat, fr1) -> 512 -> 512 -> 3
  build_fold_A<<<cdiv((long)rowsD * 544, 256), 256, 0, stream>>>(feat, fr1, 3, bufB, rowsD);
  gemm_kernel<544, 512, 4, 1><<<dim3(gD, 8), 128, gmLds(544, 4), stream>>>(
      bufB, rowsD, cM, f2w1p, nullptr, f2b1, bufA, 512);
  gemm_kernel<512, 512, 4, 1><<<dim3(gD, 8), 128, gmLds(512, 4), stream>>>(
      bufA, rowsD, cM, f2w2p, nullptr, f2b2, bufB, 512);
  gemm_kernel<512, 16, 1, 2><<<dim3(gD, 1), 128, gmLds(512, 1), stream>>>(
      bufB, rowsD, cM, f2w3p, nullptr, f2b3, outp, 3);
}
